// PaiNN_39256001085875
// MI455X (gfx1250) — matrix operand order repaired and checked by reading, not compiled, not run on hardware
//
#include <hip/hip_runtime.h>
#include <hip/hip_bf16.h>
#include <math.h>

#define N_ATOMS 15000
#define N_EDGES 300000
#define FDIM 128
#define NLAYER 3
#define N_RBF 20
#define CUTOFF 5.0f
#define MAX_Z 100

typedef __attribute__((ext_vector_type(16))) __bf16 v16bf;
typedef __attribute__((ext_vector_type(8))) float v8f;
typedef __attribute__((ext_vector_type(4))) unsigned int v4u;

#define LDQ 132
#define LCTX 264

// packed-weight per-layer offsets (uint16 elements)
#define OFS_W1 0
#define OFS_W2 16384
#define OFS_MIX 65536
#define OFS_IW1 98304
#define OFS_IW2 131072
#define LAYER_STRIDE 180224

__device__ __forceinline__ unsigned short f2bf(float f) {
  unsigned u = __float_as_uint(f);
  u += 0x7FFFu + ((u >> 16) & 1u);   // round-to-nearest-even
  return (unsigned short)(u >> 16);
}

__device__ __forceinline__ float silu_f(float x) { return x / (1.0f + __expf(-x)); }

__device__ __forceinline__ v8f zero8() {
  v8f z = {0.f, 0.f, 0.f, 0.f, 0.f, 0.f, 0.f, 0.f};
  return z;
}

__device__ __forceinline__ v8f wmma_bf16(v16bf a, v16bf b, v8f c) {
  return __builtin_amdgcn_wmma_f32_16x16x32_bf16(false, a, false, b, (short)0, c,
                                                 false, false);
}

// A-fragment (16x32 bf16): lane<16 -> K {0..7,16..23}, lane>=16 -> K {8..15,24..31}
__device__ __forceinline__ v16bf make_afrag(const float* __restrict__ row, int kb, int half) {
  v16bf a;
#pragma unroll
  for (int i = 0; i < 8; ++i) a[i] = (__bf16)row[kb + half * 8 + i];
#pragma unroll
  for (int i = 0; i < 8; ++i) a[8 + i] = (__bf16)row[kb + 16 + half * 8 + i];
  return a;
}

// B-fragment: 16 contiguous bf16 (32B) per lane from the pre-packed layout
__device__ __forceinline__ v16bf load_bfrag(const unsigned short* __restrict__ p) {
  union { v16bf v; v4u u[2]; } f;
  const v4u* q = (const v4u*)p;
  f.u[0] = q[0];
  f.u[1] = q[1];
  return f.v;
}

// ---------------------------------------------------------------------------
// init: q <- embedding[Z], mu <- 0, dmu <- 0
// ---------------------------------------------------------------------------
__global__ void init_kernel(const int* __restrict__ Z, const float* __restrict__ emb,
                            float* __restrict__ q, float* __restrict__ mu,
                            float* __restrict__ dmu) {
  long t = (long)blockIdx.x * blockDim.x + threadIdx.x;
  const long qsz = (long)N_ATOMS * FDIM;
  const long musz = (long)N_ATOMS * 3 * FDIM;
  if (t < qsz) {
    int a = (int)(t >> 7), c = (int)(t & 127);
    int z = Z[a];
    z = (z < 0) ? 0 : ((z >= MAX_Z) ? MAX_Z - 1 : z);
    q[t] = emb[z * FDIM + c];
  } else if (t < qsz + musz) {
    mu[t - qsz] = 0.0f;
  } else if (t < qsz + 2 * musz) {
    dmu[t - qsz - musz] = 0.0f;
  }
}

// ---------------------------------------------------------------------------
// pack: convert all weight matrices to bf16 WMMA B-fragment layout
//   element t = ((kt*ntiles + nt)*32 + lane)*16 + i
//   k = kt*32 + (lane<16 ? 0 : 8) + (i<8 ? i : 8 + i) ; n = nt*16 + (lane&15)
// ---------------------------------------------------------------------------
__global__ void pack_kernel(const float* __restrict__ w1, const float* __restrict__ w2,
                            const float* __restrict__ mw, const float* __restrict__ iw1,
                            const float* __restrict__ iw2, unsigned short* __restrict__ wpack) {
  int lm = blockIdx.y;
  int layer = lm / 5, mid = lm % 5;
  const float* src;
  int K, N;
  unsigned dofs;
  switch (mid) {
    case 0: src = w1 + (size_t)layer * 128 * 128; K = 128; N = 128; dofs = OFS_W1; break;
    case 1: src = w2 + (size_t)layer * 128 * 384; K = 128; N = 384; dofs = OFS_W2; break;
    case 2: src = mw + (size_t)layer * 128 * 256; K = 128; N = 256; dofs = OFS_MIX; break;
    case 3: src = iw1 + (size_t)layer * 256 * 128; K = 256; N = 128; dofs = OFS_IW1; break;
    default: src = iw2 + (size_t)layer * 128 * 384; K = 128; N = 384; dofs = OFS_IW2; break;
  }
  int total = K * N;
  int t = blockIdx.x * 256 + threadIdx.x;
  if (t >= total) return;
  unsigned short* dst = wpack + (size_t)layer * LAYER_STRIDE + dofs;
  int i = t & 15;
  int lane = (t >> 4) & 31;
  int ktnt = t >> 9;
  int ntiles = N >> 4;
  int nt = ktnt % ntiles;
  int kt = ktnt / ntiles;
  int k = kt * 32 + ((lane < 16) ? 0 : 8) + ((i < 8) ? i : 8 + i);
  int nn = nt * 16 + (lane & 15);
  dst[t] = f2bf(src[(size_t)k * N + nn]);
}

// ---------------------------------------------------------------------------
// edge geometry: dir, fcut, phi*fcut
// ---------------------------------------------------------------------------
__global__ void edge_kernel(const float* __restrict__ r_ij, float* __restrict__ dirfc,
                            float* __restrict__ phiF) {
  int e = blockIdx.x * 256 + threadIdx.x;
  if (e >= N_EDGES) return;
  float x = r_ij[3 * e + 0], y = r_ij[3 * e + 1], z = r_ij[3 * e + 2];
  float d = sqrtf(x * x + y * y + z * z);
  float inv = 1.0f / d;
  float fc = (d < CUTOFF) ? 0.5f * (__cosf(d * (3.14159265358979323846f / CUTOFF)) + 1.0f)
                          : 0.0f;
  dirfc[4 * e + 0] = x * inv;
  dirfc[4 * e + 1] = y * inv;
  dirfc[4 * e + 2] = z * inv;
  dirfc[4 * e + 3] = fc;
  const float width = CUTOFF / (float)(N_RBF - 1);
  const float coeff = -0.5f / (width * width);
#pragma unroll
  for (int k = 0; k < N_RBF; ++k) {
    float t = d - (float)k * width;
    phiF[(size_t)e * N_RBF + k] = __expf(coeff * t * t) * fc;
  }
}

// ---------------------------------------------------------------------------
// inter: x_atom = silu(q@W1+b1)@W2+b2  -- one wave per 16-atom tile, WMMA bf16,
// q tile staged through LDS with CDNA5 async global->LDS loads.
// ---------------------------------------------------------------------------
__global__ __launch_bounds__(32) void inter_kernel(
    const float* __restrict__ qg, const unsigned short* __restrict__ w1p,
    const unsigned short* __restrict__ w2p, const float* __restrict__ b1,
    const float* __restrict__ b2, float* __restrict__ x_atom) {
  __shared__ float sq[16][LDQ];
  __shared__ float sh[16][LDQ];
  const int lane = (int)threadIdx.x;
  const int m = lane & 15;
  const int half = lane >> 4;
  const int a0 = (int)blockIdx.x * 16;

  // async-stage 16x128 f32 q tile into LDS (16 x b128 per lane-wavefront)
#pragma unroll
  for (int t = 0; t < 16; ++t) {
    int idx4 = t * 32 + lane;
    int row = idx4 >> 5, c4 = idx4 & 31;
    int arow = a0 + row;
    if (arow > N_ATOMS - 1) arow = N_ATOMS - 1;
    const float* gp = qg + (size_t)arow * FDIM + c4 * 4;
    unsigned lofs = (unsigned)(unsigned long long)(&sq[row][c4 * 4]);
    asm volatile("global_load_async_to_lds_b128 %0, %1, off" ::"v"(lofs), "v"(gp)
                 : "memory");
  }
  asm volatile("s_wait_asynccnt 0" ::: "memory");
  __syncthreads();

  // h = silu(q @ W1 + b1)
  for (int nt = 0; nt < 8; ++nt) {
    v8f acc = zero8();
#pragma unroll
    for (int kt = 0; kt < 4; ++kt) {
      v16bf a = make_afrag(&sq[m][0], kt * 32, half);
      acc = wmma_bf16(a, load_bfrag(w1p + ((size_t)((kt * 8 + nt) * 32 + lane)) * 16), acc);
    }
    int col = nt * 16 + m;
    float bias = b1[col];
#pragma unroll
    for (int r = 0; r < 8; ++r) sh[r + half * 8][col] = silu_f(acc[r] + bias);
  }
  __syncthreads();

  // x = h @ W2 + b2  -> x_atom
  for (int nt = 0; nt < 24; ++nt) {
    v8f acc = zero8();
#pragma unroll
    for (int kt = 0; kt < 4; ++kt) {
      v16bf a = make_afrag(&sh[m][0], kt * 32, half);
      acc = wmma_bf16(a, load_bfrag(w2p + ((size_t)((kt * 24 + nt) * 32 + lane)) * 16), acc);
    }
    int col = nt * 16 + m;
    float bias = b2[col];
#pragma unroll
    for (int r = 0; r < 8; ++r) {
      int arow = a0 + r + half * 8;
      if (arow < N_ATOMS) x_atom[(size_t)arow * 384 + col] = acc[r] + bias;
    }
  }
}

// ---------------------------------------------------------------------------
// message: per edge, filter = phiF@Wl + fc*b; dq -> atomic into q,
// dmu -> atomic into separate dmu buffer (preserves muj read-before-update).
// 128 threads per edge, 2 edges per block.
// ---------------------------------------------------------------------------
__global__ __launch_bounds__(256) void message_kernel(
    int l, const float* __restrict__ phiF, const float* __restrict__ dirfc,
    const float* __restrict__ filt_W, const float* __restrict__ filt_b,
    const float* __restrict__ x_atom, const int* __restrict__ idx_i,
    const int* __restrict__ idx_j, const float* __restrict__ mu,
    float* __restrict__ q, float* __restrict__ dmu) {
  __shared__ float sphi[2][N_RBF];
  __shared__ float sdir[2][4];
  int sub = (int)threadIdx.x >> 7;
  int c = (int)threadIdx.x & 127;
  int e = (int)blockIdx.x * 2 + sub;
  bool valid = (e < N_EDGES);
  if (valid && c < N_RBF) sphi[sub][c] = phiF[(size_t)e * N_RBF + c];
  if (valid && c >= 64 && c < 68) sdir[sub][c - 64] = dirfc[4 * e + (c - 64)];
  __syncthreads();
  if (!valid) return;

  int j = idx_j[e], i = idx_i[e];
  const float* wcol = filt_W + l * 384 + c;  // filt_W row stride = 1152
  float f0 = 0.f, f1 = 0.f, f2 = 0.f;
#pragma unroll
  for (int r = 0; r < N_RBF; ++r) {
    float p = sphi[sub][r];
    const float* wr = wcol + (size_t)r * 1152;
    f0 += p * wr[0];
    f1 += p * wr[128];
    f2 += p * wr[256];
  }
  float fc = sdir[sub][3];
  f0 += fc * filt_b[l * 384 + c];
  f1 += fc * filt_b[l * 384 + 128 + c];
  f2 += fc * filt_b[l * 384 + 256 + c];

  const float* xr = x_atom + (size_t)j * 384;
  float dq = f0 * xr[c];
  float dmuR = f1 * xr[128 + c];
  float dmumu = f2 * xr[256 + c];
  atomicAdd(&q[(size_t)i * FDIM + c], dq);
#pragma unroll
  for (int d = 0; d < 3; ++d) {
    float muj = mu[(size_t)j * 384 + (size_t)d * 128 + c];
    atomicAdd(&dmu[(size_t)i * 384 + (size_t)d * 128 + c],
              dmuR * sdir[sub][d] + dmumu * muj);
  }
}

// mu += dmu ; dmu = 0 (for next layer)
__global__ void apply_dmu_kernel(float* __restrict__ mu, float* __restrict__ dmu) {
  int t = (int)blockIdx.x * 256 + (int)threadIdx.x;
  if (t < N_ATOMS * 3 * FDIM) {
    mu[t] += dmu[t];
    dmu[t] = 0.0f;
  }
}

// ---------------------------------------------------------------------------
// mixintra: mu_mix = mu@mixW+b -> (V,W); norms+dots; h=silu(ctx@iW1+b);
// x=h@iW2+b; q += x0 + x2*dot; mu += x1*W. One wave per 16-atom tile.
// ---------------------------------------------------------------------------
__global__ __launch_bounds__(32) void mixintra_kernel(
    float* __restrict__ qg, float* __restrict__ mug,
    const unsigned short* __restrict__ mixp, const float* __restrict__ mixb,
    const unsigned short* __restrict__ iw1p, const float* __restrict__ ib1,
    const unsigned short* __restrict__ iw2p, const float* __restrict__ ib2) {
  __shared__ float sW[3][16][LDQ];
  __shared__ float sctx[16][LCTX];  // [q(0:128) | mu_Vn(128:256)], later q-accum
  __shared__ float ssd[16][LDQ];    // sum_d V*W
  __shared__ float sh[16][LDQ];
  const int lane = (int)threadIdx.x;
  const int m = lane & 15;
  const int half = lane >> 4;
  const int a0 = (int)blockIdx.x * 16;

  for (int t = 0; t < 64; ++t) {
    int idx = t * 32 + lane;
    int row = idx >> 7, c = idx & 127;
    ssd[row][c] = 0.0f;
    sctx[row][128 + c] = 0.0f;
  }
  for (int t = 0; t < 16; ++t) {
    int idx4 = t * 32 + lane;
    int row = idx4 >> 5, c4 = idx4 & 31;
    int arow = a0 + row;
    if (arow > N_ATOMS - 1) arow = N_ATOMS - 1;
    const float* gp = qg + (size_t)arow * FDIM + c4 * 4;
    sctx[row][c4 * 4 + 0] = gp[0];
    sctx[row][c4 * 4 + 1] = gp[1];
    sctx[row][c4 * 4 + 2] = gp[2];
    sctx[row][c4 * 4 + 3] = gp[3];
  }
  __syncthreads();

  // mix GEMM per spatial dim d: V = cols 0..127, W = cols 128..255
  int arowA = a0 + m;
  if (arowA > N_ATOMS - 1) arowA = N_ATOMS - 1;
  for (int d = 0; d < 3; ++d) {
    const float* murow = mug + (size_t)arowA * 384 + (size_t)d * 128;
    v16bf afr[4];
#pragma unroll
    for (int kt = 0; kt < 4; ++kt) afr[kt] = make_afrag(murow, kt * 32, half);
    for (int nt = 0; nt < 8; ++nt) {
      v8f accV = zero8(), accW = zero8();
#pragma unroll
      for (int kt = 0; kt < 4; ++kt) {
        accV = wmma_bf16(afr[kt],
                         load_bfrag(mixp + ((size_t)((kt * 16 + nt) * 32 + lane)) * 16), accV);
        accW = wmma_bf16(
            afr[kt], load_bfrag(mixp + ((size_t)((kt * 16 + nt + 8) * 32 + lane)) * 16), accW);
      }
      int col = nt * 16 + m;
      float bV = mixb[col], bW = mixb[128 + col];
#pragma unroll
      for (int r = 0; r < 8; ++r) {
        int row = r + half * 8;
        float V = accV[r] + bV;
        float W = accW[r] + bW;
        sW[d][row][col] = W;
        sctx[row][128 + col] += V * V;
        ssd[row][col] += V * W;
      }
    }
  }
  __syncthreads();
  for (int t = 0; t < 64; ++t) {
    int idx = t * 32 + lane;
    int row = idx >> 7, c = idx & 127;
    sctx[row][128 + c] = sqrtf(sctx[row][128 + c]);  // mu_Vn
  }
  __syncthreads();

  // h = silu(ctx @ iW1 + b1), K = 256
  for (int nt = 0; nt < 8; ++nt) {
    v8f acc = zero8();
#pragma unroll
    for (int kt = 0; kt < 8; ++kt) {
      v16bf a = make_afrag(&sctx[m][0], kt * 32, half);
      acc = wmma_bf16(a, load_bfrag(iw1p + ((size_t)((kt * 8 + nt) * 32 + lane)) * 16), acc);
    }
    int col = nt * 16 + m;
    float bias = ib1[col];
#pragma unroll
    for (int r = 0; r < 8; ++r) sh[r + half * 8][col] = silu_f(acc[r] + bias);
  }
  __syncthreads();

  // x = h @ iW2 + b2 ; apply the three output segments
  for (int nt = 0; nt < 24; ++nt) {
    v8f acc = zero8();
#pragma unroll
    for (int kt = 0; kt < 4; ++kt) {
      v16bf a = make_afrag(&sh[m][0], kt * 32, half);
      acc = wmma_bf16(a, load_bfrag(iw2p + ((size_t)((kt * 24 + nt) * 32 + lane)) * 16), acc);
    }
    int col = nt * 16 + m;
    float bias = ib2[col];
    int seg = nt >> 3;
    int c = (nt & 7) * 16 + m;
#pragma unroll
    for (int r = 0; r < 8; ++r) {
      int row = r + half * 8;
      float xv = acc[r] + bias;
      if (seg == 0) {
        sctx[row][c] += xv;  // q_old + dq_intra
      } else if (seg == 1) {
        int arow = a0 + row;
        if (arow < N_ATOMS) {
#pragma unroll
          for (int d = 0; d < 3; ++d) {
            size_t o = (size_t)arow * 384 + (size_t)d * 128 + c;
            mug[o] += xv * sW[d][row][c];  // dmu_intra
          }
        }
      } else {
        sctx[row][c] += xv * ssd[row][c];  // dqmu_intra
      }
    }
  }
  __syncthreads();
  for (int t = 0; t < 64; ++t) {
    int idx = t * 32 + lane;
    int row = idx >> 7, c = idx & 127;
    int arow = a0 + row;
    if (arow < N_ATOMS) qg[(size_t)arow * FDIM + c] = sctx[row][c];
  }
}

// ---------------------------------------------------------------------------
extern "C" void kernel_launch(void* const* d_in, const int* in_sizes, int n_in,
                              void* d_out, int out_size, void* d_ws, size_t ws_size,
                              hipStream_t stream) {
  (void)in_sizes; (void)n_in; (void)out_size;
  const float* r_ij = (const float*)d_in[0];
  const float* embedding = (const float*)d_in[1];
  const float* filt_W = (const float*)d_in[2];
  const float* filt_b = (const float*)d_in[3];
  const float* inter_W1 = (const float*)d_in[4];
  const float* inter_b1 = (const float*)d_in[5];
  const float* inter_W2 = (const float*)d_in[6];
  const float* inter_b2 = (const float*)d_in[7];
  const float* mix_W = (const float*)d_in[8];
  const float* mix_b = (const float*)d_in[9];
  const float* intra_W1 = (const float*)d_in[10];
  const float* intra_b1 = (const float*)d_in[11];
  const float* intra_W2 = (const float*)d_in[12];
  const float* intra_b2 = (const float*)d_in[13];
  const int* Z = (const int*)d_in[14];
  const int* idx_i = (const int*)d_in[15];
  const int* idx_j = (const int*)d_in[16];

  float* q = (float*)d_out;                       // N_ATOMS*F
  float* mu = q + (size_t)N_ATOMS * FDIM;         // N_ATOMS*3*F

  // workspace layout
  float* x_atom = (float*)d_ws;                              // 5.76M f32
  float* dirfc = x_atom + (size_t)N_ATOMS * 384;             // 1.2M f32
  float* phiF = dirfc + (size_t)N_EDGES * 4;                 // 6.0M f32
  float* dmu = phiF + (size_t)N_EDGES * N_RBF;               // 5.76M f32
  unsigned short* wpack = (unsigned short*)(dmu + (size_t)N_ATOMS * 384);
  size_t need = ((size_t)N_ATOMS * 384 + (size_t)N_EDGES * 4 +
                 (size_t)N_EDGES * N_RBF + (size_t)N_ATOMS * 384) * 4 +
                (size_t)NLAYER * LAYER_STRIDE * 2;
  if (ws_size < need) return;

  {
    long total = (long)N_ATOMS * (FDIM + 3 * FDIM + 3 * FDIM);
    int blocks = (int)((total + 255) / 256);
    init_kernel<<<blocks, 256, 0, stream>>>(Z, embedding, q, mu, dmu);
  }
  {
    dim3 g(192, 15);
    pack_kernel<<<g, 256, 0, stream>>>(inter_W1, inter_W2, mix_W, intra_W1, intra_W2, wpack);
  }
  edge_kernel<<<(N_EDGES + 255) / 256, 256, 0, stream>>>(r_ij, dirfc, phiF);

  const int ntile = (N_ATOMS + 15) / 16;  // 938
  for (int l = 0; l < NLAYER; ++l) {
    const unsigned short* wl = wpack + (size_t)l * LAYER_STRIDE;
    inter_kernel<<<ntile, 32, 0, stream>>>(q, wl + OFS_W1, wl + OFS_W2,
                                           inter_b1 + l * FDIM, inter_b2 + l * 384, x_atom);
    message_kernel<<<N_EDGES / 2, 256, 0, stream>>>(l, phiF, dirfc, filt_W, filt_b, x_atom,
                                                    idx_i, idx_j, mu, q, dmu);
    apply_dmu_kernel<<<(N_ATOMS * 384 + 255) / 256, 256, 0, stream>>>(mu, dmu);
    mixintra_kernel<<<ntile, 32, 0, stream>>>(q, mu, wl + OFS_MIX, mix_b + l * 256,
                                              wl + OFS_IW1, intra_b1 + l * FDIM,
                                              wl + OFS_IW2, intra_b2 + l * 384);
  }
}
